// RelationalMemory_28097676050880
// MI455X (gfx1250) — matrix operand order repaired and checked by reading, not compiled, not run on hardware
//
#include <hip/hip_runtime.h>
#include <hip/hip_bf16.h>
#include <math.h>

// Problem dims (fixed by the reference)
#define BDIM 2048
#define NN   200
#define RR   256
#define EE   256
#define HH   512
#define FILLV (-10000.0f)
#define LN_EPS 1e-5f
#define CHROWS 50           // TDM chunk = 50 rows (4 chunks of 50 = 200)

typedef __attribute__((ext_vector_type(16))) __bf16       v16bf;
typedef __attribute__((ext_vector_type(8)))  float        v8f;
typedef __attribute__((ext_vector_type(4)))  unsigned int v4u;
typedef __attribute__((ext_vector_type(8)))  int          v8i_;
typedef __attribute__((ext_vector_type(4)))  int          v4i;

#if __has_builtin(__builtin_amdgcn_tensor_load_to_lds) && __has_builtin(__builtin_amdgcn_s_wait_tensorcnt)
#define HAS_TDM 1
#else
#define HAS_TDM 0
#endif

// ---------------------------------------------------------------------------
// wave32 / block(256) reduction helpers
// ---------------------------------------------------------------------------
__device__ __forceinline__ float waveSum32(float v) {
#pragma unroll
  for (int o = 16; o > 0; o >>= 1) v += __shfl_xor(v, o, 32);
  return v;
}

__device__ __forceinline__ float blockSum256(float v, float* scratch) {
  const int lane = threadIdx.x & 31, wave = threadIdx.x >> 5;
  v = waveSum32(v);
  if (lane == 0) scratch[wave] = v;
  __syncthreads();
  float r = scratch[0];
#pragma unroll
  for (int i = 1; i < 8; ++i) r += scratch[i];
  __syncthreads();
  return r;
}

__device__ __forceinline__ float blockMax256(float v, float* scratch) {
  const int lane = threadIdx.x & 31, wave = threadIdx.x >> 5;
#pragma unroll
  for (int o = 16; o > 0; o >>= 1) v = fmaxf(v, __shfl_xor(v, o, 32));
  if (lane == 0) scratch[wave] = v;
  __syncthreads();
  float r = scratch[0];
#pragma unroll
  for (int i = 1; i < 8; ++i) r = fmaxf(r, scratch[i]);
  __syncthreads();
  return r;
}

// ---------------------------------------------------------------------------
// Weight prep: f32 -> bf16, optionally transposed into B^T layout [N][K].
// ---------------------------------------------------------------------------
__global__ void f32_to_bf16_kernel(const float* __restrict__ in,
                                   __bf16* __restrict__ out, int n) {
  const int i = blockIdx.x * 256 + threadIdx.x;
  if (i < n) out[i] = (__bf16)in[i];
}

__global__ void transpose_bf16_kernel(const float* __restrict__ in /*[K][N]*/,
                                      __bf16* __restrict__ out /*[N][K]*/,
                                      int K, int N) {
  const int i = blockIdx.x * 256 + threadIdx.x;
  if (i >= K * N) return;
  const int n = i / K, k = i - n * K;
  out[i] = (__bf16)in[(size_t)k * N + n];
}

// ---------------------------------------------------------------------------
// WMMA GEMM: C[M,NDIM] = act(A[M,KDIM] @ W + bias), W pre-converted bf16 in
// B^T layout Wb[NDIM][KDIM]. ACT: 0 none, 1 sigmoid, 2 tanh.
// Block = 128 thr (4 wave32s), grid (M/16, NDIM/64); wave w owns 16x16 tile.
// ---------------------------------------------------------------------------
template <int KDIM, int NDIM, int ACT>
__global__ __launch_bounds__(128) void gemm_wmma_kernel(
    const float* __restrict__ A, const __bf16* __restrict__ Wb,
    const float* __restrict__ bias, float* __restrict__ C) {
  const int lane = threadIdx.x & 31;
  const int wave = threadIdx.x >> 5;
  const int l15  = lane & 15;
  const int hi   = lane >> 4;                 // 0: lanes 0-15, 1: lanes 16-31
  const int row0 = blockIdx.x << 4;
  const int col0 = (blockIdx.y << 6) + (wave << 4);
  const int mrow = row0 + l15;                // A-frag: M = lane%16
  const int ncol = col0 + l15;                // B/C-frag: N = lane%16

  const float*  arow = A  + (size_t)mrow * KDIM + (hi ? 16 : 0);
  const __bf16* brow = Wb + (size_t)ncol * KDIM + (hi ? 16 : 0);
  v8f acc = {};

  for (int k0 = 0; k0 < KDIM; k0 += 32) {
    // A fragment (16x32 bf16): lane-half selects K groups {0-15}/{16-31}
    float4 a0 = *(const float4*)(arow + k0);
    float4 a1 = *(const float4*)(arow + k0 + 4);
    float4 a2 = *(const float4*)(arow + k0 + 8);
    float4 a3 = *(const float4*)(arow + k0 + 12);
    v16bf af;
    af[0] = (__bf16)a0.x; af[1] = (__bf16)a0.y; af[2] = (__bf16)a0.z; af[3] = (__bf16)a0.w;
    af[4] = (__bf16)a1.x; af[5] = (__bf16)a1.y; af[6] = (__bf16)a1.z; af[7] = (__bf16)a1.w;
    af[8]  = (__bf16)a2.x; af[9]  = (__bf16)a2.y; af[10] = (__bf16)a2.z; af[11] = (__bf16)a2.w;
    af[12] = (__bf16)a3.x; af[13] = (__bf16)a3.y; af[14] = (__bf16)a3.z; af[15] = (__bf16)a3.w;

    // B fragment (32x16 bf16): one contiguous 32B load
    v16bf bf = *(const v16bf*)(brow + k0);

    acc = __builtin_amdgcn_wmma_f32_16x16x32_bf16(
        /*neg_a=*/false, af, /*neg_b=*/false, bf,
        /*c_mod=*/(short)0, acc, /*reuse_a=*/false, /*reuse_b=*/false);
  }

  const float bv = bias ? bias[ncol] : 0.0f;
  const int mbase = row0 + (hi ? 8 : 0);      // C/D: M = r + 8*(lane>=16)
#pragma unroll
  for (int r = 0; r < 8; ++r) {
    float v = acc[r] + bv;
    if (ACT == 1)      v = 1.0f / (1.0f + __expf(-v));
    else if (ACT == 2) v = tanhf(v);
    C[(size_t)(mbase + r) * NDIM + ncol] = v;
  }
}

// ---------------------------------------------------------------------------
// Fused attention: one workgroup (256 thr) per batch row b.
// TDM streams nb_rel[b] (200 KB) into LDS in 4 x 50-row chunks, software-
// pipelined against the u-dot phase via TENSORcnt; then masked softmax and
// z = attn_w @ nb_rel[b], all out of LDS. Single HBM pass over nb_rel.
// ---------------------------------------------------------------------------
__global__ __launch_bounds__(256) void attn_fused_kernel(
    const float* __restrict__ nb_rel, const float* __restrict__ delta_t,
    const unsigned char* __restrict__ hist_mask,   // jnp bool_ = 1 byte
    const float* __restrict__ u, const float* __restrict__ log_gamma,
    float* __restrict__ z) {
  extern __shared__ float lds[];
  float* tile  = lds;                 // [NN * RR]  (200 KB)
  float* sattn = lds + NN * RR;       // [256]
  float* sred  = sattn + 256;         // [64] reduction scratch (+flag)
  int*   sflag = (int*)(sred + 32);

  const int b    = blockIdx.x;
  const int tid  = threadIdx.x;
  const int lane = tid & 31;
  const int wave = tid >> 5;
  const float gamma = __expf(log_gamma[0]);

  if (tid == 0) *sflag = 0;
  if (tid >= NN) sattn[tid] = -1e30f; // pad [200,256)

  // u[b] slice held in registers: lane owns elements [8*lane, 8*lane+8)
  const float* up = u + (size_t)b * RR + lane * 8;
  const float4 u0 = *(const float4*)(up);
  const float4 u1 = *(const float4*)(up + 4);

  int any = 0;

#if HAS_TDM
  // ---- Issue all 4 chunked TDM loads from wave 0 (TENSORcnt-tracked) ------
  if (wave == 0) {
    const unsigned ldsBase = (unsigned)(uintptr_t)tile;   // flat->LDS: addr[31:0]
    const unsigned long long ga =
        (unsigned long long)(uintptr_t)(nb_rel + ((size_t)b * NN) * RR);
#pragma unroll
    for (int c = 0; c < 4; ++c) {
      const unsigned long long gaddr =
          ga + (unsigned long long)c * (CHROWS * RR * 4);
      v4u g0;
      g0[0] = 1u;                                          // count=1 (valid D#)
      g0[1] = ldsBase + (unsigned)(c * (CHROWS * RR * 4)); // lds_addr
      g0[2] = (unsigned)gaddr;                             // global_addr[31:0]
      g0[3] = (unsigned)((gaddr >> 32) & 0x1FFFFFFu) | (2u << 30); // [56:32]|type=2
      v8i_ g1;
      g1[0] = (int)(2u << 16);                  // data_size = 2 (4 bytes)
      g1[1] = (int)((unsigned)RR << 16);        // tensor_dim0 = 256
      g1[2] = (int)((unsigned)CHROWS << 16);    // tensor_dim1 = 50 (chunk rows)
      g1[3] = (int)((unsigned)RR << 16);        // tile_dim0 = 256
      g1[4] = CHROWS;                           // tile_dim1 = 50, tile_dim2 = 0
      g1[5] = RR;                               // tensor_dim0_stride = 256
      g1[6] = 0;
      g1[7] = 0;
      v4i  gz4; gz4[0] = 0; gz4[1] = 0; gz4[2] = 0; gz4[3] = 0;   // groups 2/3
      v8i_ gz8; gz8[0] = 0; gz8[1] = 0; gz8[2] = 0; gz8[3] = 0;
                gz8[4] = 0; gz8[5] = 0; gz8[6] = 0; gz8[7] = 0;
      __builtin_amdgcn_tensor_load_to_lds(g0, g1, gz4, gz4, gz8, 0);
    }
  }

  // ---- Pipelined logit phase: wait chunk c, compute dots from LDS ---------
  auto chunk_dots = [&](int c) {
    for (int n = c * CHROWS + wave; n < (c + 1) * CHROWS; n += 8) {
      const float* tp = &tile[(size_t)n * RR + lane * 8];
      float4 x0 = *(const float4*)(tp);
      float4 x1 = *(const float4*)(tp + 4);
      float p = x0.x * u0.x + x0.y * u0.y + x0.z * u0.z + x0.w * u0.w +
                x1.x * u1.x + x1.y * u1.y + x1.z * u1.z + x1.w * u1.w;
      p = waveSum32(p);
      if (lane == 0) {
        const int m = hist_mask[(size_t)b * NN + n];
        float a;
        if (m) {
          const float dt = delta_t[(size_t)b * NN + n];
          a = p * 0.0625f + fmaxf(-gamma * fmaxf(dt, 0.0f), -10.0f);
          any = 1;
        } else {
          a = FILLV;
        }
        sattn[n] = a;
      }
    }
  };
  // manual unroll: s_wait_tensorcnt needs a literal immediate
  if (wave == 0) __builtin_amdgcn_s_wait_tensorcnt(3);
  __syncthreads();
  chunk_dots(0);
  if (wave == 0) __builtin_amdgcn_s_wait_tensorcnt(2);
  __syncthreads();
  chunk_dots(1);
  if (wave == 0) __builtin_amdgcn_s_wait_tensorcnt(1);
  __syncthreads();
  chunk_dots(2);
  if (wave == 0) __builtin_amdgcn_s_wait_tensorcnt(0);
  __syncthreads();
  chunk_dots(3);
#else
  // ---- Fallback: manual global->LDS streaming fused with the dot ----------
  __syncthreads();
  for (int n = wave; n < NN; n += 8) {
    const float* rp = nb_rel + ((size_t)b * NN + n) * RR + lane * 8;
    __builtin_prefetch(rp + 8 * RR, 0, 0);
    float4 x0 = *(const float4*)(rp);
    float4 x1 = *(const float4*)(rp + 4);
    *(float4*)(&tile[(size_t)n * RR + lane * 8])     = x0;
    *(float4*)(&tile[(size_t)n * RR + lane * 8 + 4]) = x1;
    float p = x0.x * u0.x + x0.y * u0.y + x0.z * u0.z + x0.w * u0.w +
              x1.x * u1.x + x1.y * u1.y + x1.z * u1.z + x1.w * u1.w;
    p = waveSum32(p);
    if (lane == 0) {
      const int m = hist_mask[(size_t)b * NN + n];
      float a;
      if (m) {
        const float dt = delta_t[(size_t)b * NN + n];
        a = p * 0.0625f + fmaxf(-gamma * fmaxf(dt, 0.0f), -10.0f);
        any = 1;
      } else {
        a = FILLV;
      }
      sattn[n] = a;
    }
  }
#endif

  if (lane == 0 && any) atomicOr(sflag, 1);
  __syncthreads();

  // ---- Masked softmax over N ----------------------------------------------
  const bool allmask = (*sflag == 0);
  float a = sattn[tid];
  if (allmask && tid < NN) a = 0.0f;
  const float mx = blockMax256(a, sred);
  const float e  = (tid < NN) ? __expf(a - mx) : 0.0f;
  const float s  = blockSum256(e, sred);
  __syncthreads();
  sattn[tid] = e / s;
  __syncthreads();

  // ---- z[b,r] = sum_n w[n] * tile[n,r]  (bank-conflict-free over r) -------
  float acc = 0.0f;
#pragma unroll 4
  for (int n = 0; n < NN; ++n) acc += sattn[n] * tile[(size_t)n * RR + tid];
  z[(size_t)b * RR + tid] = acc;
}

// ---------------------------------------------------------------------------
// Elementwise helpers
// ---------------------------------------------------------------------------
__global__ void concat_kernel(const float* __restrict__ m_init,
                              const float* __restrict__ agg,
                              float* __restrict__ cat) {
  const int idx = blockIdx.x * blockDim.x + threadIdx.x;
  if (idx >= BDIM * 2 * EE) return;
  const int row = idx >> 9;
  const int c   = idx & 511;
  cat[idx] = (c < EE) ? m_init[(size_t)row * EE + c]
                      : agg[(size_t)row * EE + (c - EE)];
}

__global__ void combine_kernel(const float* __restrict__ gate,
                               const float* __restrict__ tt,
                               const float* __restrict__ m_init,
                               float* __restrict__ m_out) {
  const int idx = blockIdx.x * blockDim.x + threadIdx.x;
  if (idx >= BDIM * EE) return;
  const float g = gate[idx];
  m_out[idx] = g * tt[idx] + (1.0f - g) * m_init[idx];
}

__global__ __launch_bounds__(256) void ln_kernel(
    const float* __restrict__ y, const float* __restrict__ g,
    const float* __restrict__ bb, float* __restrict__ out) {
  __shared__ float sred[8];
  const int row = blockIdx.x;
  const int t   = threadIdx.x;
  const float v0 = y[(size_t)row * HH + t];
  const float v1 = y[(size_t)row * HH + t + 256];
  const float mu = blockSum256(v0 + v1, sred) * (1.0f / HH);
  const float d0 = v0 - mu, d1 = v1 - mu;
  const float var = blockSum256(d0 * d0 + d1 * d1, sred) * (1.0f / HH);
  const float rs = rsqrtf(var + LN_EPS);
  out[(size_t)row * HH + t]       = d0 * rs * g[t]       + bb[t];
  out[(size_t)row * HH + t + 256] = d1 * rs * g[t + 256] + bb[t + 256];
}

// ---------------------------------------------------------------------------
// Orchestration
// ---------------------------------------------------------------------------
extern "C" void kernel_launch(void* const* d_in, const int* in_sizes, int n_in,
                              void* d_out, int out_size, void* d_ws, size_t ws_size,
                              hipStream_t stream) {
  const float* nb_rel  = (const float*)d_in[0];
  const float* delta_t = (const float*)d_in[1];
  const float* r_emb   = (const float*)d_in[2];
  const unsigned char* hist_mask = (const unsigned char*)d_in[3];
  const float* Wi = (const float*)d_in[4];
  const float* bi = (const float*)d_in[5];
  const float* Wm = (const float*)d_in[6];
  const float* Wk = (const float*)d_in[7];
  const float* lg = (const float*)d_in[8];   // log_gamma scalar
  const float* Wg = (const float*)d_in[9];
  const float* bg = (const float*)d_in[10];
  const float* Wt = (const float*)d_in[11];
  const float* bt = (const float*)d_in[12];
  const float* Wo = (const float*)d_in[13];
  const float* bo = (const float*)d_in[14];
  const float* lng = (const float*)d_in[15];
  const float* lnb = (const float*)d_in[16];
  float* out = (float*)d_out;

  // Workspace carving: f32 activations, then bf16 weight copies
  float* ws = (float*)d_ws;
  const size_t SBE = (size_t)BDIM * EE;      // 524288
  float* m_init = ws;           ws += SBE;
  float* q_key  = ws;           ws += SBE;
  float* u      = ws;           ws += SBE;   // [B,R]
  float* z      = ws;           ws += SBE;
  float* agg    = ws;           ws += SBE;
  float* gate   = ws;           ws += SBE;
  float* tt     = ws;           ws += SBE;
  float* m_out  = ws;           ws += SBE;
  float* cat    = ws;           ws += (size_t)BDIM * 2 * EE;
  float* y      = ws;           ws += (size_t)BDIM * HH;

  __bf16* wb = (__bf16*)ws;
  const size_t W66 = 256 * 256, W131 = 512 * 256;
  __bf16* Wi_t   = wb;  wb += W66;   // [E][R]   (B^T of Wi[R,E])
  __bf16* Wk_t   = wb;  wb += W66;   // [E][E]
  __bf16* Wm_row = wb;  wb += W66;   // Wm[R,E] as-is == B^T for u-GEMM
  __bf16* Wm_t   = wb;  wb += W66;   // [E][R]
  __bf16* Wg_t   = wb;  wb += W131;  // [256][512]
  __bf16* Wt_t   = wb;  wb += W66;   // [E][E]
  __bf16* Wo_t   = wb;               // [512][256]

  // ---- Weight prep (one-shot, tiny) ----
  transpose_bf16_kernel<<<(256 * 256 + 255) / 256, 256, 0, stream>>>(Wi, Wi_t, 256, 256);
  transpose_bf16_kernel<<<(256 * 256 + 255) / 256, 256, 0, stream>>>(Wk, Wk_t, 256, 256);
  f32_to_bf16_kernel   <<<(256 * 256 + 255) / 256, 256, 0, stream>>>(Wm, Wm_row, 256 * 256);
  transpose_bf16_kernel<<<(256 * 256 + 255) / 256, 256, 0, stream>>>(Wm, Wm_t, 256, 256);
  transpose_bf16_kernel<<<(512 * 256 + 255) / 256, 256, 0, stream>>>(Wg, Wg_t, 512, 256);
  transpose_bf16_kernel<<<(256 * 256 + 255) / 256, 256, 0, stream>>>(Wt, Wt_t, 256, 256);
  transpose_bf16_kernel<<<(256 * 512 + 255) / 256, 256, 0, stream>>>(Wo, Wo_t, 256, 512);

  const dim3 blk(128);
  const dim3 g256(BDIM / 16, EE / 64);       // (128, 4)
  const dim3 g512(BDIM / 16, HH / 64);       // (128, 8)

  // m_init = r_emb @ Wi + bi ; q_key = m_init @ Wk ; u = q_key @ Wm^T
  gemm_wmma_kernel<256, 256, 0><<<g256, blk, 0, stream>>>(r_emb,  Wi_t,   bi,      m_init);
  gemm_wmma_kernel<256, 256, 0><<<g256, blk, 0, stream>>>(m_init, Wk_t,   nullptr, q_key);
  gemm_wmma_kernel<256, 256, 0><<<g256, blk, 0, stream>>>(q_key,  Wm_row, nullptr, u);

  // Fused single-pass attention (TDM-streamed) -> z = attn_w @ nb_rel
  const size_t shmem = (size_t)(NN * RR + 256 + 64) * sizeof(float);  // ~202 KB
  attn_fused_kernel<<<dim3(BDIM), dim3(256), shmem, stream>>>(
      nb_rel, delta_t, hist_mask, u, lg, z);

  // agg = z @ Wm ; gate = sigmoid([m_init,agg]@Wg+bg) ; tt = tanh(agg@Wt+bt)
  gemm_wmma_kernel<256, 256, 0><<<g256, blk, 0, stream>>>(z, Wm_t, nullptr, agg);
  concat_kernel<<<(BDIM * 512 + 255) / 256, 256, 0, stream>>>(m_init, agg, cat);
  gemm_wmma_kernel<512, 256, 1><<<g256, blk, 0, stream>>>(cat, Wg_t, bg, gate);
  gemm_wmma_kernel<256, 256, 2><<<g256, blk, 0, stream>>>(agg, Wt_t, bt, tt);
  // m_out = gate*tt + (1-gate)*m_init ; y = m_out @ Wo + bo ; out = LN(y)
  combine_kernel<<<(BDIM * EE + 255) / 256, 256, 0, stream>>>(gate, tt, m_init, m_out);
  gemm_wmma_kernel<256, 512, 0><<<g512, blk, 0, stream>>>(m_out, Wo_t, bo, y);
  ln_kernel<<<BDIM, 256, 0, stream>>>(y, lng, lnb, out);
}
